// Allegro_62362925137936
// MI455X (gfx1250) — matrix operand order repaired and checked by reading, not compiled, not run on hardware
//
#include <hip/hip_runtime.h>
#include <hip/hip_bf16.h>
#include <math.h>

#define N_NODES 20000
#define N_EDGES 320000

typedef __attribute__((ext_vector_type(16))) _Float16     v16h;
typedef __attribute__((ext_vector_type(8)))  float        v8f;
typedef __attribute__((ext_vector_type(4)))  unsigned int u32x4;
typedef __attribute__((ext_vector_type(4)))  int          i32x4;
typedef __attribute__((ext_vector_type(8)))  int          i32x8;

#if __has_builtin(__builtin_amdgcn_tensor_load_to_lds) && __has_builtin(__builtin_amdgcn_s_wait_tensorcnt)
#define HAVE_TDM 1
#else
#define HAVE_TDM 0
#endif

union Frag { v16h h; uint4 u[2]; };

__device__ inline unsigned short f2h(float x) {
    union { _Float16 h; unsigned short u; } t; t.h = (_Float16)x; return t.u;
}

__device__ __forceinline__ void loadA(Frag& f, const _Float16* ap) {
    f.u[0] = *(const uint4*)(ap);
    f.u[1] = *(const uint4*)(ap + 8);
}

// ---------------------------------------------------------------------------
// Generic WMMA GEMM: C[M x N] = act( (A[M x K] @ Bsw[K x N]) * scale )
//   M = N_EDGES (multiple of 128), K multiple of 32, N = NT*16.
//   8 waves / block; each wave owns a 16-row stripe and all N tiles.
//   Bsw is ALREADY in WMMA fragment order (dword i = frag[ks][nt][lane][v],
//   produced by k_cvtw), so staging to LDS is a flat contiguous copy — done
//   with the Tensor Data Mover (1-D descriptor) when available.
//   Fragment maps (A/B k order is this code's choice):
//     A (16x32 f16): lane L (m=L&15, half=L>>4), VGPR v packs
//       K = half*16 + 2v  -> two contiguous 16B global loads.
//     B (32x16 f16): lane L (n=L&15, half=L>>4), VGPR v packs K = half*16+2v
//       -> 8 contiguous dwords in LDS (two ds_load_b128).
//     C/D: m = v + 8*half, n = L&15.
//   A is software-pipelined with two NAMED fragments (a0/a1) and a ks-loop
//   unrolled by 2 — no runtime-indexed frag array (that spills to scratch).
// ---------------------------------------------------------------------------
template<int NT>
__global__ __launch_bounds__(256) void gemm_k(
    const _Float16* __restrict__ A, int lda,
    const unsigned int* __restrict__ Bsw,
    int K, float scale, int act,
    _Float16* __restrict__ C16, float* __restrict__ C32, int ldc)
{
    extern __shared__ unsigned int ldsB[];
    const int tid  = threadIdx.x;
    const int lane = tid & 31;
    const int wave = tid >> 5;
    const int ksteps  = K >> 5;
    const int totaldw = (K * (NT * 16)) >> 1;

#if HAVE_TDM
    if (wave == 0) {
        unsigned long long ga = (unsigned long long)(uintptr_t)Bsw;
        unsigned int nd = (unsigned int)totaldw;           // dwords to move
        u32x4 g0;
        g0[0] = 1u;                                        // count=1, user desc
        g0[1] = 0u;                                        // lds_addr = 0 (dyn-LDS base)
        g0[2] = (unsigned int)ga;                          // global_addr[31:0]
        g0[3] = (unsigned int)((ga >> 32) & 0x01FFFFFFu)   // global_addr[56:32]
              | (2u << 30);                                // type = 2 ("image")
        i32x8 g1 = (i32x8)0;
        g1[0] = (int)(2u << 16);                           // data_size = 4 bytes
        g1[1] = (int)((nd & 0xFFFFu) << 16);               // tensor_dim0[15:0]  @bit48
        g1[2] = (int)((nd >> 16) & 0xFFFFu);               // tensor_dim0[31:16]
        g1[3] = (int)((nd & 0xFFFFu) << 16);               // tile_dim0          @bit112
        g1[5] = (int)nd;                                   // tensor_dim0_stride[31:0]
        i32x4 z4 = (i32x4)0;
#if __has_include(<hip/amd_detail/amd_gfx1250_TDM.h>)
        i32x8 z8 = (i32x8)0;
        __builtin_amdgcn_tensor_load_to_lds(g0, g1, z4, z4, z8, 0);
#else
        __builtin_amdgcn_tensor_load_to_lds(g0, g1, z4, z4, 0);
#endif
        __builtin_amdgcn_s_wait_tensorcnt(0);
    }
#else
    for (int i = tid; i < totaldw; i += 256) ldsB[i] = Bsw[i];
#endif

    const int rowbase = blockIdx.x * 128 + wave * 16;
    const int half = lane >> 4;
    const int row  = rowbase + (lane & 15);
    const _Float16* ap0 = A + (size_t)row * lda + half * 16;

    v8f acc[NT] = {};
    auto mmstep = [&](const Frag& a, int ks) {
        const unsigned int* bp = ldsB + (size_t)ks * (NT * 256) + (lane << 3);
#pragma unroll
        for (int nt = 0; nt < NT; ++nt) {
            Frag b;
            b.u[0] = *(const uint4*)(bp + nt * 256);
            b.u[1] = *(const uint4*)(bp + nt * 256 + 4);
            acc[nt] = __builtin_amdgcn_wmma_f32_16x16x32_f16(
                false, a.h, false, b.h, (short)0, acc[nt], false, false);
        }
    };

    Frag a0, a1;
    loadA(a0, ap0);                        // preload ks=0 before the barrier
    __syncthreads();

    int ks = 0;
    for (; ks + 2 <= ksteps; ks += 2) {    // pipelined pairs: named regs only
        loadA(a1, ap0 + (ks + 1) * 32);
        __builtin_prefetch(ap0 + (ks + 1) * 32 + 32, 0, 1);   // global_prefetch_b8
        mmstep(a0, ks);
        if (ks + 2 < ksteps) loadA(a0, ap0 + (ks + 2) * 32);
        mmstep(a1, ks + 1);
    }
    if (ks < ksteps) mmstep(a0, ks);       // odd-ksteps tail

#pragma unroll
    for (int nt = 0; nt < NT; ++nt) {
        int n = (nt << 4) + (lane & 15);
#pragma unroll
        for (int v = 0; v < 8; ++v) {
            size_t m = (size_t)rowbase + v + (half << 3);
            float val = acc[nt][v] * scale;
            if (act) val = val / (1.0f + __expf(-val));   // SiLU
            if (C32) C32[m * ldc + n] = val;
            if (C16) C16[m * ldc + n] = (_Float16)val;
        }
    }
}

// ---------------------------------------------------------------------------
// weight f32 -> f16, zero-padded in K, emitted in WMMA fragment order:
//   dword i = frag[ks][nt][lane][v];  kk = ks*32 + (lane>>4)*16 + 2v,
//   n = nt*16 + (lane&15);  dword packs (B[kk][n], B[kk+1][n]).
// ---------------------------------------------------------------------------
__global__ void k_cvtw(const float* __restrict__ src, int srcK, int N,
                       unsigned int* __restrict__ dst, int dstK)
{
    int i = blockIdx.x * 256 + threadIdx.x;
    int total = (dstK * N) >> 1;
    if (i >= total) return;
    int NT = N >> 4;
    int ks = i / (NT * 256);
    int r  = i - ks * (NT * 256);
    int nt = r >> 8;
    int l  = (r >> 3) & 31;
    int v  = r & 7;
    int kk = ks * 32 + ((l >> 4) << 4) + (v << 1);
    int n  = (nt << 4) + (l & 15);
    unsigned int lo = f2h((kk     < srcK) ? src[(size_t)kk * N + n]       : 0.0f);
    unsigned int hi = f2h((kk + 1 < srcK) ? src[(size_t)(kk + 1) * N + n] : 0.0f);
    dst[i] = lo | (hi << 16);
}

// bessel basis normalization constants via trapz (matches numpy reference)
__global__ void k_bconsts(float* c)
{
    int k = threadIdx.x;
    if (k >= 8) return;
    const double h  = (1.0 - 1e-9) / 999.0;
    const double kk = 3.14159265358979323846 * (double)(k + 1);
    double mu = 0.0, fprev = 0.0;
    for (int i = 0; i < 1000; ++i) {
        double r = 1e-9 + h * i;
        double f = 1.4142135623730951 * sin(kk * r) / r;
        if (i > 0) mu += 0.5 * (fprev + f) * h;
        fprev = f;
    }
    double var = 0.0; fprev = 0.0;
    for (int i = 0; i < 1000; ++i) {
        double r = 1e-9 + h * i;
        double f = 1.4142135623730951 * sin(kk * r) / r;
        double g = (f - mu) * (f - mu);
        if (i > 0) var += 0.5 * (fprev + g) * h;
        fprev = g;
    }
    c[k]     = (float)mu;
    c[8 + k] = (float)sqrt(var);
}

// per-edge: bessel+attrs -> x0p (padded to 64), env, Y[9]
__global__ void k_feat(const float* __restrict__ ev, const float* __restrict__ attrs,
                       const int* __restrict__ esrc, const int* __restrict__ edst,
                       const float* __restrict__ bc,
                       _Float16* __restrict__ x0p, float* __restrict__ env,
                       float* __restrict__ Y)
{
    int e = blockIdx.x * 256 + threadIdx.x;
    if (e >= N_EDGES) return;
    float dx = ev[e * 3], dy = ev[e * 3 + 1], dz = ev[e * 3 + 2];
    float d2 = dx * dx + dy * dy + dz * dz;
    float d  = sqrtf(d2 == 0.f ? 1.f : d2);
    _Float16* xr = x0p + (size_t)e * 64;
    const float s2 = 1.41421356237309515f;
    const float PI = 3.14159265358979323846f;
#pragma unroll
    for (int k = 0; k < 8; ++k) {
        float b = s2 * sinf(PI * (float)(k + 1) * d) / d;
        xr[k] = (_Float16)((b - bc[k]) / bc[8 + k]);
    }
    int s = esrc[e], t = edst[e];
#pragma unroll
    for (int j = 0; j < 16; ++j) {
        xr[8 + j]  = (_Float16)attrs[s * 16 + j];
        xr[24 + j] = (_Float16)attrs[t * 16 + j];
    }
    // envelope, p=6: 1 - 28 d^6 + 48 d^7 - 21 d^8
    float d3 = d * d * d, d6 = d3 * d3, d7 = d6 * d, d8 = d7 * d;
    env[e] = 1.f - 28.f * d6 + 48.f * d7 - 21.f * d8;
    float inv = 1.f / d;
    float x = dx * inv, y = dy * inv, z = dz * inv;
    const float s3 = 1.73205080756887729f, s15 = 3.87298334620741689f, s5 = 2.23606797749978970f;
    float* Ye = Y + (size_t)e * 9;
    Ye[0] = 1.f;
    Ye[1] = s3 * x;  Ye[2] = s3 * y;  Ye[3] = s3 * z;
    Ye[4] = s15 * x * y;  Ye[5] = s15 * y * z;
    Ye[6] = 0.5f * s5 * (3.f * z * z - 1.f);
    Ye[7] = s15 * x * z;  Ye[8] = 0.5f * s15 * (x * x - y * y);
}

__global__ void k_envscale(float* __restrict__ x32, _Float16* __restrict__ x16,
                           const float* __restrict__ env)
{
    int i = blockIdx.x * 256 + threadIdx.x;
    if (i >= N_EDGES * 128) return;
    float v = x32[i] * env[i >> 7];
    x32[i] = v; x16[i] = (_Float16)v;
}

__global__ void k_v1init(const float* __restrict__ V0, const float* __restrict__ Y,
                         float* __restrict__ vx, float* __restrict__ vy, float* __restrict__ vz)
{
    int i = blockIdx.x * 256 + threadIdx.x;
    if (i >= N_EDGES * 32) return;
    int e = i >> 5;
    float w = V0[i];
    const float* Ye = Y + (size_t)e * 9;
    vx[i] = w * Ye[1]; vy[i] = w * Ye[2]; vz[i] = w * Ye[3];
}

__global__ void k_scatter(const float* __restrict__ w2, const float* __restrict__ Y,
                          const int* __restrict__ esrc, float* __restrict__ agg)
{
    int i = blockIdx.x * 256 + threadIdx.x;
    if (i >= N_EDGES * 32) return;
    int e = i >> 5, c = i & 31;
    float v = w2[i];
    const float* Ye = Y + (size_t)e * 9;
    float* a = agg + (size_t)esrc[e] * 288 + c * 9;
#pragma unroll
    for (int j = 0; j < 9; ++j) atomicAdd(a + j, v * Ye[j]);
}

// gather agg, compute scalars/vectors, build latent-MLP input + Vm planes
__global__ void k_gather(const float* __restrict__ agg, const int* __restrict__ esrc,
                         const float* __restrict__ v1x, const float* __restrict__ v1y,
                         const float* __restrict__ v1z, const float* __restrict__ V0,
                         const _Float16* __restrict__ x16,
                         _Float16* __restrict__ xin, int ldxin,
                         _Float16* __restrict__ vmx, _Float16* __restrict__ vmy,
                         _Float16* __restrict__ vmz, int ldm)
{
    int i = blockIdx.x * 256 + threadIdx.x;
    if (i >= N_EDGES * 32) return;
    int e = i >> 5, c = i & 31;
    const float* a = agg + (size_t)esrc[e] * 288 + c * 9;
    float A0  = a[0] * 0.25f;                        // / sqrt(NUM_NEIGH)
    float A1x = a[1] * 0.25f, A1y = a[2] * 0.25f, A1z = a[3] * 0.25f;
    float axy = a[4] * 0.25f, ayz = a[5] * 0.25f, az2 = a[6] * 0.25f;
    float axz = a[7] * 0.25f, ax2 = a[8] * 0.25f;
    size_t ec = (size_t)e * 32 + c;
    float vx = v1x[ec], vy = v1y[ec], vz = v1z[ec];
    float sc1 = (A1x * vx + A1y * vy + A1z * vz) * 0.57735026918962576f;
    // tp_2e_1o
    const float s15h = 1.93649167310370844f, s5h = 1.11803398874989485f, s5 = 2.23606797749978970f;
    float mxx = -s5h * az2 + s15h * ax2;
    float myy = -s5h * az2 - s15h * ax2;
    float mzz =  s5  * az2;
    float mxy = s15h * axy, mxz = s15h * axz, myz = s15h * ayz;
    const float itp = 0.28284271247461901f;          // 1/sqrt(12.5)
    float tx = (mxx * vx + mxy * vy + mxz * vz) * itp;
    float ty = (mxy * vx + myy * vy + myz * vz) * itp;
    float tz = (mxz * vx + myz * vy + mzz * vz) * itp;

    _Float16* xr = xin + (size_t)e * ldxin;
    const _Float16* xs = x16 + (size_t)e * 128;
#pragma unroll
    for (int j = 0; j < 4; ++j) xr[c * 4 + j] = xs[c * 4 + j];   // copy x into cols 0..127

    size_t vm = (size_t)e * ldm;
    if (V0) {  // layer 1: scal = [A0*V0, <A1,V1>/sqrt3]; vecs = [A0*V1, A1*V0, tp]
        float v0 = V0[ec];
        xr[128 + c * 2 + 0] = (_Float16)(A0 * v0);
        xr[128 + c * 2 + 1] = (_Float16)sc1;
        vmx[vm + c]      = (_Float16)(A0 * vx);
        vmy[vm + c]      = (_Float16)(A0 * vy);
        vmz[vm + c]      = (_Float16)(A0 * vz);
        vmx[vm + 32 + c] = (_Float16)(A1x * v0);
        vmy[vm + 32 + c] = (_Float16)(A1y * v0);
        vmz[vm + 32 + c] = (_Float16)(A1z * v0);
        vmx[vm + 64 + c] = (_Float16)tx;
        vmy[vm + 64 + c] = (_Float16)ty;
        vmz[vm + 64 + c] = (_Float16)tz;
    } else {   // layers 2,3
        xr[128 + c] = (_Float16)sc1;
        vmx[vm + c]      = (_Float16)(A0 * vx);
        vmy[vm + c]      = (_Float16)(A0 * vy);
        vmz[vm + c]      = (_Float16)(A0 * vz);
        vmx[vm + 32 + c] = (_Float16)tx;
        vmy[vm + 32 + c] = (_Float16)ty;
        vmz[vm + 32 + c] = (_Float16)tz;
    }
}

__global__ void k_residual(float* __restrict__ x32, _Float16* __restrict__ x16,
                           const _Float16* __restrict__ y, const float* __restrict__ env)
{
    int i = blockIdx.x * 256 + threadIdx.x;
    if (i >= N_EDGES * 128) return;
    float v = (x32[i] + 0.5f * env[i >> 7] * (float)y[i]) * 0.89442719099991586f;
    x32[i] = v; x16[i] = (_Float16)v;
}

__global__ void k_out(const _Float16* __restrict__ xf, const float* __restrict__ wout,
                      float* __restrict__ out)
{
    int e = blockIdx.x * 256 + threadIdx.x;
    if (e >= N_EDGES) return;
    const _Float16* xr = xf + (size_t)e * 128;
    float s = 0.f;
#pragma unroll 8
    for (int f = 0; f < 128; ++f) s += (float)xr[f] * wout[f];
    out[e] = s * 0.08838834764831845f;
}

// ---------------------------------------------------------------------------
// host side
// ---------------------------------------------------------------------------
static void gemm(hipStream_t s, const _Float16* A, int lda, const unsigned int* Bsw,
                 int K, int N, float scale, int act,
                 _Float16* C16, float* C32, int ldc)
{
    dim3 g(N_EDGES / 128), b(256);
    size_t lds = (size_t)K * N * 2;
    switch (N / 16) {
        case 1: gemm_k<1><<<g, b, lds, s>>>(A, lda, Bsw, K, scale, act, C16, C32, ldc); break;
        case 2: gemm_k<2><<<g, b, lds, s>>>(A, lda, Bsw, K, scale, act, C16, C32, ldc); break;
        case 4: gemm_k<4><<<g, b, lds, s>>>(A, lda, Bsw, K, scale, act, C16, C32, ldc); break;
        case 8: gemm_k<8><<<g, b, lds, s>>>(A, lda, Bsw, K, scale, act, C16, C32, ldc); break;
    }
}

extern "C" void kernel_launch(void* const* d_in, const int* in_sizes, int n_in,
                              void* d_out, int out_size, void* d_ws, size_t ws_size,
                              hipStream_t stream)
{
    const int E = N_EDGES;
    const float* node_attrs = (const float*)d_in[0];
    const float* edge_vecs  = (const float*)d_in[1];
    const float* w_emb[4]   = {(const float*)d_in[2], (const float*)d_in[3],
                               (const float*)d_in[4], (const float*)d_in[5]};
    const float* w_w[4]     = {(const float*)d_in[6], (const float*)d_in[7],
                               (const float*)d_in[12], (const float*)d_in[17]};
    const float* lat[3][3]  = {{(const float*)d_in[8],  (const float*)d_in[9],  (const float*)d_in[10]},
                               {(const float*)d_in[13], (const float*)d_in[14], (const float*)d_in[15]},
                               {(const float*)d_in[18], (const float*)d_in[19], (const float*)d_in[20]}};
    const float* lin[3]     = {(const float*)d_in[11], (const float*)d_in[16], (const float*)d_in[21]};
    const float* w_f        = (const float*)d_in[22];
    const float* w_out      = (const float*)d_in[23];
    const int*   esrc       = (const int*)d_in[24];
    const int*   edst       = (const int*)d_in[25];

    // --- workspace carve (bump allocator, 256B aligned) ---
    char* base = (char*)d_ws; size_t off = 0;
    auto alloc = [&](size_t bytes) -> void* {
        void* p = base + off; off = (off + bytes + 255) & ~(size_t)255; return p;
    };
    float*     bconsts = (float*)alloc(64);
    float*     envb    = (float*)alloc((size_t)E * 4);
    float*     Yb      = (float*)alloc((size_t)E * 36);
    float*     x32     = (float*)alloc((size_t)E * 512);
    _Float16*  x16     = (_Float16*)alloc((size_t)E * 256);
    float*     V0b     = (float*)alloc((size_t)E * 128);
    float*     v1p[6];
    for (int i = 0; i < 6; ++i) v1p[i] = (float*)alloc((size_t)E * 128);
    float*     agg     = (float*)alloc((size_t)N_NODES * 1152);
    float*     w2b     = (float*)alloc((size_t)E * 128);
    _Float16*  xin     = (_Float16*)alloc((size_t)E * 384);   // up to E x 192 f16
    _Float16*  vm[3];
    for (int t = 0; t < 3; ++t) vm[t] = (_Float16*)alloc((size_t)E * 192);  // E x 96 f16
    _Float16*  t1      = (_Float16*)alloc((size_t)E * 256);
    _Float16*  t2      = (_Float16*)alloc((size_t)E * 256);
    _Float16*  yb      = (_Float16*)alloc((size_t)E * 256);
    unsigned int* wh   = (unsigned int*)alloc(1 << 21);       // swizzled f16 weights
    // phase-safe aliases (embedding happens before xin/vm/t1 are used)
    _Float16* x0p = xin;      // E x 64
    _Float16* h1  = vm[0];    // E x 32 (padded)
    _Float16* h2  = vm[1];    // E x 32
    _Float16* h3  = t1;       // E x 64

    // --- fragment-swizzled f16 weight copies (with K padding), dword units ---
    size_t woff = 0;
    auto walloc = [&](size_t ndw) -> unsigned int* {
        unsigned int* p = wh + woff; woff += (ndw + 63) & ~(size_t)63; return p;
    };
    auto cvt = [&](const float* src, int srcK, int N, unsigned int* dst, int dstK) {
        int tot = (dstK * N) >> 1;
        k_cvtw<<<(tot + 255) / 256, 256, 0, stream>>>(src, srcK, N, dst, dstK);
    };
    unsigned int* emb0h = walloc(64 * 16 / 2);
    unsigned int* emb1h = walloc(32 * 32 / 2);
    unsigned int* emb2h = walloc(32 * 64 / 2);
    unsigned int* emb3h = walloc(64 * 128 / 2);
    unsigned int* wwh[4];  for (int i = 0; i < 4; ++i) wwh[i] = walloc(128 * 32 / 2);
    unsigned int* lath[3][3];
    for (int l = 0; l < 3; ++l) {
        lath[l][0] = walloc((size_t)(l == 0 ? 192 : 160) * 128 / 2);
        lath[l][1] = walloc(128 * 128 / 2);
        lath[l][2] = walloc(128 * 128 / 2);
    }
    unsigned int* linh[3];
    for (int l = 0; l < 3; ++l) linh[l] = walloc((size_t)(l == 0 ? 96 : 64) * 32 / 2);
    unsigned int* wfh = walloc(128 * 128 / 2);

    k_bconsts<<<1, 8, 0, stream>>>(bconsts);
    cvt(w_emb[0], 40, 16, emb0h, 64);
    cvt(w_emb[1], 16, 32, emb1h, 32);
    cvt(w_emb[2], 32, 64, emb2h, 32);
    cvt(w_emb[3], 64, 128, emb3h, 64);
    for (int i = 0; i < 4; ++i) cvt(w_w[i], 128, 32, wwh[i], 128);
    for (int l = 0; l < 3; ++l) {
        int Kin = (l == 0) ? 192 : 160;
        cvt(lat[l][0], Kin, 128, lath[l][0], Kin);
        cvt(lat[l][1], 128, 128, lath[l][1], 128);
        cvt(lat[l][2], 128, 128, lath[l][2], 128);
        cvt(lin[l], (l == 0) ? 96 : 64, 32, linh[l], (l == 0) ? 96 : 64);
    }
    cvt(w_f, 128, 128, wfh, 128);

    // zero padded activation buffers (every call: ws is poisoned once pre-timing)
    hipMemsetAsync(x0p, 0, (size_t)E * 128, stream);   // E x 64 f16
    hipMemsetAsync(h1,  0, (size_t)E * 64,  stream);   // E x 32 f16

    const float s128 = 0.08838834764831845f;
    const int gE    = (E + 255) / 256;
    const int gE32  = (E * 32 + 255) / 256;
    const int gE128 = (E * 128 + 255) / 256;

    // featurize + embedding MLP (WMMA chain) + env scale
    k_feat<<<gE, 256, 0, stream>>>(edge_vecs, node_attrs, esrc, edst, bconsts, x0p, envb, Yb);
    gemm(stream, x0p, 64, emb0h, 64, 16,  0.15811388300841897f, 1, h1, nullptr, 32);
    gemm(stream, h1,  32, emb1h, 32, 32,  0.25f,                1, h2, nullptr, 32);
    gemm(stream, h2,  32, emb2h, 32, 64,  0.17677669529663687f, 1, h3, nullptr, 64);
    gemm(stream, h3,  64, emb3h, 64, 128, 0.125f,               0, nullptr, x32, 128);
    k_envscale<<<gE128, 256, 0, stream>>>(x32, x16, envb);

    // V0 = x @ w_w0 / sqrt(F); V1 = V0 (x) Y1
    gemm(stream, x16, 128, wwh[0], 128, 32, s128, 0, nullptr, V0b, 32);
    k_v1init<<<gE32, 256, 0, stream>>>(V0b, Yb, v1p[0], v1p[1], v1p[2]);

    float* v1cur[3] = {v1p[0], v1p[1], v1p[2]};
    float* v1nxt[3] = {v1p[3], v1p[4], v1p[5]};
    for (int l = 0; l < 3; ++l) {
        int Kin = (l == 0) ? 192 : 160;
        int m   = (l == 0) ? 96 : 64;
        float sK = (l == 0) ? 0.07216878364870322f : 0.07905694150420949f;
        float sm = (l == 0) ? 0.10206207261596575f : 0.125f;

        gemm(stream, x16, 128, wwh[l + 1], 128, 32, s128, 0, nullptr, w2b, 32);
        hipMemsetAsync(agg, 0, (size_t)N_NODES * 1152, stream);
        k_scatter<<<gE32, 256, 0, stream>>>(w2b, Yb, esrc, agg);
        k_gather<<<gE32, 256, 0, stream>>>(agg, esrc, v1cur[0], v1cur[1], v1cur[2],
                                           (l == 0) ? V0b : nullptr, x16,
                                           xin, Kin, vm[0], vm[1], vm[2], m);
        // latent MLP
        gemm(stream, xin, Kin, lath[l][0], Kin, 128, sK,   1, t1, nullptr, 128);
        gemm(stream, t1,  128, lath[l][1], 128, 128, s128, 1, t2, nullptr, 128);
        gemm(stream, t2,  128, lath[l][2], 128, 128, s128, 0, yb, nullptr, 128);
        // Vnew = Vm @ lin / sqrt(m), per component
        for (int t = 0; t < 3; ++t)
            gemm(stream, vm[t], m, linh[l], m, 32, sm, 0, nullptr, v1nxt[t], 32);
        k_residual<<<gE128, 256, 0, stream>>>(x32, x16, yb, envb);
        for (int t = 0; t < 3; ++t) { float* tmp = v1cur[t]; v1cur[t] = v1nxt[t]; v1nxt[t] = tmp; }
    }

    // head: x @ w_f / sqrt(128), then dot with w_out / sqrt(128)
    gemm(stream, x16, 128, wfh, 128, 128, s128, 0, t2, nullptr, 128);
    k_out<<<gE, 256, 0, stream>>>(t2, w_out, (float*)d_out);
}
